// Simulator_15401752724176
// MI455X (gfx1250) — compile-verified, matrix operand order checked by reading, not run on hardware
//
#include <hip/hip_runtime.h>
#include <hip/hip_bf16.h>

typedef __attribute__((ext_vector_type(16))) _Float16 v16h;
typedef __attribute__((ext_vector_type(8)))  _Float16 v8h;
typedef __attribute__((ext_vector_type(8)))  float    v8f;

#define LDA 136   // LDS row stride in halves (128 + 8 pad -> bank spread)

struct MlpParams {
  const float* W[8];
  const float* b[8];
};

// ---------------------------------------------------------------------------
// One 32x64 wave-tile GEMM layer: D = relu(A[32,K] * W^T[K,64] + bias)
// Two M-subtiles share each B-tile -> fewer LDS B loads per WMMA.
// Operand layouts per CDNA5 ISA 16-bit WMMA maps (lane = row/col, two
// 16-byte K-chunks per lane, lanes 16-31 take the K+8 / K+24 chunks).
// ---------------------------------------------------------------------------
template<int KTILES>
__device__ inline void gemm_layer(const _Float16* inP, _Float16* outP,
                                  const _Float16* wbuf, const float* bias,
                                  int lane, int mbase, int nbase)
{
  const int l15 = lane & 15;
  const int h   = lane >> 4;
  const _Float16* aRow0 = inP + (mbase + l15) * LDA + h * 8;
  const _Float16* aRow1 = aRow0 + 16 * LDA;
#pragma unroll
  for (int nt = 0; nt < 4; ++nt) {
    const int n0 = nbase + nt * 16;
    const float bv = bias[n0 + l15];            // bias depends only on N-column
    v8f c0 = {bv, bv, bv, bv, bv, bv, bv, bv};
    v8f c1 = {bv, bv, bv, bv, bv, bv, bv, bv};
    const _Float16* bRow = wbuf + (n0 + l15) * LDA + h * 8;
#pragma unroll
    for (int kt = 0; kt < KTILES; ++kt) {
      const int KB = kt * 32;
      v8h blo = *(const v8h*)(bRow + KB);
      v8h bhi = *(const v8h*)(bRow + KB + 16);
      v8h a0l = *(const v8h*)(aRow0 + KB);
      v8h a0h = *(const v8h*)(aRow0 + KB + 16);
      v8h a1l = *(const v8h*)(aRow1 + KB);
      v8h a1h = *(const v8h*)(aRow1 + KB + 16);
      v16h B, A0, A1;
#pragma unroll
      for (int i = 0; i < 8; ++i) {
        B[i]  = blo[i]; B[i + 8]  = bhi[i];
        A0[i] = a0l[i]; A0[i + 8] = a0h[i];
        A1[i] = a1l[i]; A1[i + 8] = a1h[i];
      }
      c0 = __builtin_amdgcn_wmma_f32_16x16x32_f16(
          false, A0, false, B, (short)0, c0, false, false);
      c1 = __builtin_amdgcn_wmma_f32_16x16x32_f16(
          false, A1, false, B, (short)0, c1, false, false);
    }
    // C/D layout: VGPR r -> M = r + 8*h ; N = n0 + (lane&15)
#pragma unroll
    for (int r = 0; r < 8; ++r) {
      float v0 = fmaxf(c0[r], 0.0f);  // all 7 WMMA layers are followed by relu
      float v1 = fmaxf(c1[r], 0.0f);
      outP[(mbase + r + h * 8) * LDA + n0 + l15]      = (_Float16)v0;
      outP[(mbase + 16 + r + h * 8) * LDA + n0 + l15] = (_Float16)v1;
    }
  }
}

__device__ inline void load_weights_lds(_Float16* wbuf, const _Float16* gsrc, int tid)
{
  // 128 rows x 128 halves, copied as 16-byte chunks into padded LDS
  for (int c = tid; c < 2048; c += 256) {
    const int row = c >> 4;
    const int col = (c & 15) << 3;
    *(v8h*)&wbuf[row * LDA + col] = *(const v8h*)&gsrc[row * 128 + col];
  }
}

// ---------------------------------------------------------------------------
// Fused per-step kernel: edge feature build -> 7 WMMA layers -> dot -> scatter
// ---------------------------------------------------------------------------
__global__ __launch_bounds__(256) void fused_step(
    const float* __restrict__ coords, const float* __restrict__ node_f,
    const float* __restrict__ res_num,
    const int* __restrict__ snd, const int* __restrict__ rcv,
    const _Float16* __restrict__ wt, MlpParams P,
    float* __restrict__ tforce, const float* __restrict__ radius_p, int Eper)
{
  __shared__ __align__(16) _Float16 actA[128 * LDA];
  __shared__ __align__(16) _Float16 actB[128 * LDA];
  __shared__ __align__(16) _Float16 wbuf[128 * LDA];
  __shared__ float ndL[128 * 3];
  __shared__ int   ridx[128];

  const int tid = threadIdx.x;

  // ---- stage 128 edge messages as f16 rows (cols 0..63, zero padded) ----
  if (tid < 128) {
    const long e = (long)blockIdx.x * 128 + tid;
    _Float16* m = &actA[tid * LDA];
    float nx = 0.f, ny = 0.f, nz = 0.f;
    int ri = 0;
    if (e < Eper) {
      const int s = snd[e];
      const int r = rcv[e];
      const float dx = coords[s * 3 + 0] - coords[r * 3 + 0];
      const float dy = coords[s * 3 + 1] - coords[r * 3 + 1];
      const float dz = coords[s * 3 + 2] - coords[r * 3 + 2];
      const float radius = *radius_p;
      const float d = sqrtf(dx * dx + dy * dy + dz * dz) / radius;
      const float mden = fmaxf(d, 0.01f);
      nx = dx / mden; ny = dy / mden; nz = dz / mden;
      const float seq = fminf(fabsf(res_num[s] - res_num[r]) * 0.2f, 1.0f);
#pragma unroll
      for (int k = 0; k < 24; ++k) {
        m[k]      = (_Float16)node_f[s * 24 + k];
        m[24 + k] = (_Float16)node_f[r * 24 + k];
      }
      m[48] = (_Float16)d;
      m[49] = (_Float16)seq;
#pragma unroll
      for (int k = 50; k < 64; ++k) m[k] = (_Float16)0.f;
      ri = r;
    } else {
#pragma unroll
      for (int k = 0; k < 64; ++k) m[k] = (_Float16)0.f;
    }
    ndL[tid * 3 + 0] = nx; ndL[tid * 3 + 1] = ny; ndL[tid * 3 + 2] = nz;
    ridx[tid] = ri;
  }

  const int lane  = tid & 31;
  const int wave  = tid >> 5;
  const int mbase = (wave & 3) * 32;   // 4 M-groups x 32 rows = 128 rows
  const int nbase = (wave >> 2) * 64;  // 2 N-groups x 64 cols = 128 cols

  // ---- layer 0: K = 64 (2 k-tiles), msg(actA) -> actB ----
  __syncthreads();
  load_weights_lds(wbuf, wt, tid);
  __builtin_prefetch(wt + 16384, 0, 1);
  __syncthreads();
  gemm_layer<2>(actA, actB, wbuf, P.b[0], lane, mbase, nbase);

  // ---- layers 1..6: K = 128 (4 k-tiles), ping-pong ----
  const _Float16* inP = actB;
  _Float16* outP = actA;
  for (int l = 1; l < 7; ++l) {
    __syncthreads();
    load_weights_lds(wbuf, wt + (size_t)l * 16384, tid);
    __builtin_prefetch(wt + (size_t)(l + 1) * 16384, 0, 1);
    __syncthreads();
    gemm_layer<4>(inP, outP, wbuf, P.b[l], lane, mbase, nbase);
    _Float16* t = (_Float16*)inP; inP = outP; outP = t;
  }
  // after 6 iterations the layer-6 output is in actB

  // ---- layer 7: [128] . w7 -> scalar, scale by norm_diff, scatter-add ----
  __syncthreads();
  if (tid < 16) {   // w7 stored as row 0 of layer-7 slot
    *(v8h*)&wbuf[tid * 8] = *(const v8h*)&wt[(size_t)7 * 16384 + tid * 8];
  }
  __syncthreads();
  if (tid < 128) {
    const long e = (long)blockIdx.x * 128 + tid;
    if (e < Eper) {
      float acc = P.b[7][0];
      const _Float16* hrow = &actB[tid * LDA];
#pragma unroll 8
      for (int k = 0; k < 128; ++k)
        acc += (float)hrow[k] * (float)wbuf[k];
      const int ri = ridx[tid];
      atomicAdd(&tforce[ri * 3 + 0], acc * ndL[tid * 3 + 0]);
      atomicAdd(&tforce[ri * 3 + 1], acc * ndL[tid * 3 + 1]);
      atomicAdd(&tforce[ri * 3 + 2], acc * ndL[tid * 3 + 2]);
    }
  }
}

// ---------------------------------------------------------------------------
// Small helper kernels
// ---------------------------------------------------------------------------
__global__ void convert_weights(MlpParams P, _Float16* wt)
{
  const int idx = blockIdx.x * blockDim.x + threadIdx.x;
  if (idx >= 8 * 128 * 128) return;
  const int l = idx >> 14;
  const int rest = idx & 16383;
  const int o = rest >> 7;      // output channel (row of W^T)
  const int i = rest & 127;     // input channel  (col of W^T)
  const int Kl[8] = {50, 128, 128, 128, 128, 128, 128, 128};
  const int Nl[8] = {128, 128, 128, 128, 128, 128, 128, 1};
  float v = 0.f;
  if (i < Kl[l] && o < Nl[l]) v = P.W[l][i * Nl[l] + o];
  wt[idx] = (_Float16)v;
}

__global__ void zero_f32(float* p, int n)
{
  const int i = blockIdx.x * blockDim.x + threadIdx.x;
  if (i < n) p[i] = 0.f;
}

__global__ void integrate(float* coords, const float* vels, const float* accs,
                          const float* dt_p, int n3)
{
  const int i = blockIdx.x * blockDim.x + threadIdx.x;
  if (i < n3) {
    const float dt = *dt_p;
    coords[i] += vels[i] * dt + 0.5f * accs[i] * dt * dt;
  }
}

__global__ void half_kick(const float* tforce, const float* masses,
                          float* vels, float* accs, const float* dt_p, int n)
{
  const int i = blockIdx.x * blockDim.x + threadIdx.x;
  if (i < n) {
    const float dt = *dt_p;
    const float invm = 1.0f / masses[i];
#pragma unroll
    for (int c = 0; c < 3; ++c) {
      const float a = tforce[i * 3 + c] * invm;
      vels[i * 3 + c] += 0.5f * (accs[i * 3 + c] + a) * dt;
      accs[i * 3 + c] = a;
    }
  }
}

// ---------------------------------------------------------------------------
// Host launcher (graph-capture safe: only kernel launches + async D2D copies)
// ---------------------------------------------------------------------------
extern "C" void kernel_launch(void* const* d_in, const int* in_sizes, int n_in,
                              void* d_out, int out_size, void* d_ws, size_t ws_size,
                              hipStream_t stream)
{
  if (n_in < 25) return;
  const float* coords0 = (const float*)d_in[0];
  const float* node_f  = (const float*)d_in[1];
  const float* res_num = (const float*)d_in[2];
  const float* masses  = (const float*)d_in[3];
  const float* vels0   = (const float*)d_in[4];
  const int*   senders = (const int*)d_in[5];
  const int*   recvrs  = (const int*)d_in[6];
  MlpParams P;
  for (int i = 0; i < 8; ++i) {
    P.W[i] = (const float*)d_in[7 + i];
    P.b[i] = (const float*)d_in[15 + i];
  }
  const float* radius_p = (const float*)d_in[23];
  const float* dt_p     = (const float*)d_in[24];

  const int STEPS = 2;
  const int Etot = in_sizes[5];
  const int Eper = Etot / STEPS;
  const int N3   = in_sizes[0];
  const int N    = N3 / 3;

  // workspace carve (256-byte aligned regions)
  size_t off = 0;
  auto carve = [&](size_t bytes) {
    size_t o = off;
    off = (off + bytes + 255) & ~(size_t)255;
    return o;
  };
  char* w = (char*)d_ws;
  float*     coordsW = (float*)(w + carve((size_t)N3 * 4));
  float*     velsW   = (float*)(w + carve((size_t)N3 * 4));
  float*     accsW   = (float*)(w + carve((size_t)N3 * 4));
  float*     tforce  = (float*)(w + carve((size_t)N3 * 4));
  _Float16*  wt      = (_Float16*)(w + carve((size_t)8 * 128 * 128 * 2));
  (void)ws_size;

  // initial state
  hipMemcpyAsync(coordsW, coords0, (size_t)N3 * 4, hipMemcpyDeviceToDevice, stream);
  hipMemcpyAsync(velsW,   vels0,   (size_t)N3 * 4, hipMemcpyDeviceToDevice, stream);
  zero_f32<<<(N3 + 255) / 256, 256, 0, stream>>>(accsW, N3);
  convert_weights<<<(8 * 128 * 128 + 255) / 256, 256, 0, stream>>>(P, wt);

  const int eblocks = (Eper + 127) / 128;
  for (int s = 0; s < STEPS; ++s) {
    zero_f32<<<(N3 + 255) / 256, 256, 0, stream>>>(tforce, N3);
    integrate<<<(N3 + 255) / 256, 256, 0, stream>>>(coordsW, velsW, accsW, dt_p, N3);
    fused_step<<<eblocks, 256, 0, stream>>>(
        coordsW, node_f, res_num,
        senders + (size_t)s * Eper, recvrs + (size_t)s * Eper,
        wt, P, tforce, radius_p, Eper);
    half_kick<<<(N + 255) / 256, 256, 0, stream>>>(tforce, masses, velsW, accsW, dt_p, N);
  }

  // final coords -> output
  hipMemcpyAsync(d_out, coordsW, (size_t)N3 * 4, hipMemcpyDeviceToDevice, stream);
  (void)out_size;
}
